// SMoELayer_14370960573220
// MI455X (gfx1250) — matrix operand order repaired and checked by reading, not compiled, not run on hardware
//
#include <hip/hip_runtime.h>

// ---------------------------------------------------------------------------
// Threshold-gated MoE with LoRA experts, MI455X (gfx1250).
//   w[t,e]           = renorm(threshold(softmax(x @ gate_w + gate_b)))
//   wdown[t,e*64+r]  = w[t,e] * (x[t,:] @ lora_A[e,:,r])     bf16 WMMA GEMM
//   out[t,c]         = 0.5 * wdown[t,:] @ lora_B_flat[:,c]   bf16 WMMA GEMM
// lora_B [E,R,C] flattens to row-major [512,1024] directly.
// Pipeline: double-buffered LDS, v_perm-packed bf16 dword LDS stores, async
// global->LDS (b128) for the pre-converted bf16 A operand of GEMM 2.
// ---------------------------------------------------------------------------

typedef __bf16 bf16_t;
typedef __attribute__((ext_vector_type(8)))  __bf16 v8bf;
typedef __attribute__((ext_vector_type(16))) __bf16 v16bf;
typedef __attribute__((ext_vector_type(8)))  float  v8f;

#define T_TOK   32768
#define C_DIM   1024
#define E_EXP   8
#define N_DOWN  512      // E * R
#define THRESH  0.1f
#define SCALING 0.5f

#define TM  128          // workgroup M tile
#define TN  128          // workgroup N tile
#define TK  32           // K step (one bf16 WMMA K)
#define LDA 40           // padded LDS row stride (bf16): 80B -> 20 banks, conflict-free
#define LDB 40

// Two fp32 -> packed bf16 dword: round-half-up adds + one v_perm_b32
// (dest bytes = {a.b2, a.b3, b.b2, b.b3} -> selector 0x07060302 with s0=b,s1=a).
static __device__ __forceinline__ unsigned pk2_bf16(float a, float b) {
  union { float f; unsigned u; } x, y; x.f = a; y.f = b;
  return __builtin_amdgcn_perm(y.u + 0x8000u, x.u + 0x8000u, 0x07060302u);
}
// fp32 -> bf16 bit pattern; (u+0x8000)>>16 stored as u16 lowers to store_d16_hi.
static __device__ __forceinline__ unsigned short bf16_bits(float f) {
  union { float f; unsigned u; } v; v.f = f;
  return (unsigned short)((v.u + 0x8000u) >> 16);
}
static __device__ __forceinline__ v16bf frag16(const bf16_t* p0, const bf16_t* p1) {
  v8bf lo = *(const v8bf*)p0;
  v8bf hi = *(const v8bf*)p1;
  return __builtin_shufflevector(lo, hi, 0,1,2,3,4,5,6,7,8,9,10,11,12,13,14,15);
}

// --------------------------- gating ----------------------------------------
__global__ __launch_bounds__(256) void smoe_gate(const float* __restrict__ x,
                                                 const float* __restrict__ gw,
                                                 const float* __restrict__ gb,
                                                 float* __restrict__ wout) {
  const int gtid = blockIdx.x * 256 + threadIdx.x;
  const int tok  = gtid >> 5;
  const int lane = threadIdx.x & 31;
  const float* xr = x + (size_t)tok * C_DIM;

  float acc[E_EXP];
#pragma unroll
  for (int e = 0; e < E_EXP; ++e) acc[e] = 0.f;

  for (int c = lane; c < C_DIM; c += 32) {
    float xv = xr[c];
    const float4* g4 = (const float4*)(gw + (size_t)c * E_EXP);
    float4 g0 = g4[0], g1 = g4[1];
    acc[0] += xv * g0.x; acc[1] += xv * g0.y;
    acc[2] += xv * g0.z; acc[3] += xv * g0.w;
    acc[4] += xv * g1.x; acc[5] += xv * g1.y;
    acc[6] += xv * g1.z; acc[7] += xv * g1.w;
  }
#pragma unroll
  for (int off = 16; off > 0; off >>= 1)
#pragma unroll
    for (int e = 0; e < E_EXP; ++e)
      acc[e] += __shfl_xor(acc[e], off, 32);

  if (lane == 0) {
    float mx = -3.4e38f;
#pragma unroll
    for (int e = 0; e < E_EXP; ++e) { acc[e] += gb[e]; mx = fmaxf(mx, acc[e]); }
    float s = 0.f;
#pragma unroll
    for (int e = 0; e < E_EXP; ++e) { acc[e] = __expf(acc[e] - mx); s += acc[e]; }
    float inv = 1.f / s;
    float ws = 0.f;
#pragma unroll
    for (int e = 0; e < E_EXP; ++e) {
      float p = acc[e] * inv;
      p = (p >= THRESH) ? p : 0.f;
      acc[e] = p; ws += p;
    }
    float wn = (ws == 0.f) ? 1.f : (1.f / ws);
    float* wr = wout + (size_t)tok * E_EXP;
#pragma unroll
    for (int e = 0; e < E_EXP; ++e) wr[e] = acc[e] * wn;
  }
}

// --------------------- GEMM 1: wdown = diag(w) * (x @ A_all) ----------------
__global__ __launch_bounds__(256) void smoe_down(const float* __restrict__ x,
                                                 const float* __restrict__ loraA,
                                                 const float* __restrict__ w,
                                                 unsigned short* __restrict__ wdown) {
  __shared__ bf16_t As[2][TM * LDA];    // [m][k]
  __shared__ bf16_t Bs[2][TN * LDB];    // transposed: [n][k]
  const int tid  = threadIdx.x;
  const int lane = tid & 31;
  const int wid  = tid >> 5;
  const int wm   = wid & 3;
  const int wn   = wid >> 2;
  const int m0   = blockIdx.y * TM;
  const int n0   = blockIdx.x * TN;

  v8f acc[2][4];
#pragma unroll
  for (int mi = 0; mi < 2; ++mi)
#pragma unroll
    for (int ni = 0; ni < 4; ++ni)
#pragma unroll
      for (int i = 0; i < 8; ++i) acc[mi][ni][i] = 0.f;

  // A: thread -> (row, 16-wide k segment)
  const int ar = tid >> 1;
  const int ak = (tid & 1) << 4;
  // B: thread -> one column n, 16 consecutive k (coalesced b32 loads along n)
  const int nB   = tid & 127;
  const int kseg = (tid >> 7) << 4;                 // 0 or 16
  const int eB   = (n0 + nB) >> 6;
  const int rB   = (n0 + nB) & 63;
  const float* bbase = loraA + ((size_t)eB * C_DIM) * 64 + rB;

  // WMMA fragment addressing (CDNA5 16-bit layouts)
  const int arow  = wm * 32 + (lane & 15);
  const int akoff = (lane < 16) ? 0 : 8;
  const int bcol  = wn * 64 + (lane & 15);
  const int bkoff = (lane < 16) ? 0 : 8;

  float av[16], bv[16];
  auto load_tile = [&](int k0) {
    const float4* ap = (const float4*)(x + (size_t)(m0 + ar) * C_DIM + k0 + ak);
#pragma unroll
    for (int i = 0; i < 4; ++i) ((float4*)av)[i] = ap[i];
#pragma unroll
    for (int i = 0; i < 16; ++i) bv[i] = bbase[(size_t)(k0 + kseg + i) * 64];
  };
  auto store_tile = [&](int buf) {
    unsigned ua[8], ub[8];
#pragma unroll
    for (int i = 0; i < 8; ++i) ua[i] = pk2_bf16(av[2 * i], av[2 * i + 1]);
#pragma unroll
    for (int i = 0; i < 8; ++i) ub[i] = pk2_bf16(bv[2 * i], bv[2 * i + 1]);
    *(uint4*)(&As[buf][ar * LDA + ak])       = ((uint4*)ua)[0];
    *(uint4*)(&As[buf][ar * LDA + ak + 8])   = ((uint4*)ua)[1];
    *(uint4*)(&Bs[buf][nB * LDB + kseg])     = ((uint4*)ub)[0];
    *(uint4*)(&Bs[buf][nB * LDB + kseg + 8]) = ((uint4*)ub)[1];
  };
  auto compute = [&](int buf) {
    v16bf afr[2], bfr[4];
#pragma unroll
    for (int mi = 0; mi < 2; ++mi) {
      const bf16_t* p = &As[buf][(arow + mi * 16) * LDA + akoff];
      afr[mi] = frag16(p, p + 16);
    }
#pragma unroll
    for (int ni = 0; ni < 4; ++ni) {
      const bf16_t* p = &Bs[buf][(bcol + ni * 16) * LDB + bkoff];
      bfr[ni] = frag16(p, p + 16);
    }
#pragma unroll
    for (int mi = 0; mi < 2; ++mi)
#pragma unroll
      for (int ni = 0; ni < 4; ++ni)
        acc[mi][ni] = __builtin_amdgcn_wmma_f32_16x16x32_bf16(
            false, afr[mi], false, bfr[ni], (short)0, acc[mi][ni], false, false);
  };

  const int NIT = C_DIM / TK;
  load_tile(0);
  store_tile(0);
  __syncthreads();
#pragma unroll 1
  for (int it = 0; it < NIT; ++it) {
    const int buf = it & 1;
    if (it + 1 < NIT) load_tile((it + 1) * TK);   // prefetch next tile
    compute(buf);
    if (it + 1 < NIT) {
      store_tile(buf ^ 1);
      __syncthreads();
    }
  }

  // Epilogue: scale rows by gate weight of this wave's (single) expert -> bf16.
  const int e = (n0 + wn * 64) >> 6;
#pragma unroll
  for (int mi = 0; mi < 2; ++mi) {
    const int rbase = m0 + wm * 32 + mi * 16 + ((lane >> 4) << 3);
    float wv[8];
#pragma unroll
    for (int i = 0; i < 8; ++i) wv[i] = w[(size_t)(rbase + i) * E_EXP + e];
    const int colb = n0 + wn * 64 + (lane & 15);
#pragma unroll
    for (int ni = 0; ni < 4; ++ni) {
      const int col = colb + ni * 16;
#pragma unroll
      for (int i = 0; i < 8; ++i)
        wdown[(size_t)(rbase + i) * N_DOWN + col] = bf16_bits(acc[mi][ni][i] * wv[i]);
    }
  }
}

// ------------------- GEMM 2: out = 0.5 * wdown @ B_flat ---------------------
__global__ __launch_bounds__(256) void smoe_up(const bf16_t* __restrict__ wdown,
                                               const float* __restrict__ loraB,
                                               float* __restrict__ out) {
  __shared__ bf16_t As[2][TM * LDA];
  __shared__ bf16_t Bs[2][TN * LDB];
  const int tid  = threadIdx.x;
  const int lane = tid & 31;
  const int wid  = tid >> 5;
  const int wm   = wid & 3;
  const int wn   = wid >> 2;
  const int m0   = blockIdx.y * TM;
  const int n0   = blockIdx.x * TN;

  v8f acc[2][4];
#pragma unroll
  for (int mi = 0; mi < 2; ++mi)
#pragma unroll
    for (int ni = 0; ni < 4; ++ni)
#pragma unroll
      for (int i = 0; i < 8; ++i) acc[mi][ni][i] = 0.f;

  const int ar = tid >> 1;
  const int ak = (tid & 1) << 4;
  const int nB   = tid & 127;
  const int kseg = (tid >> 7) << 4;

  const int arow  = wm * 32 + (lane & 15);
  const int akoff = (lane < 16) ? 0 : 8;
  const int bcol  = wn * 64 + (lane & 15);
  const int bkoff = (lane < 16) ? 0 : 8;

  float bv[16];
  // A tile is already bf16 in global: move it with gfx1250 async global->LDS.
  // INST_OFFSET is added to both the LDS and global address (ISA 8.2/§4.4).
  auto issue_a_async = [&](int k0, int buf) {
    unsigned lds0 = (unsigned)(uintptr_t)(&As[buf][ar * LDA + ak]);
    unsigned long long g0 =
        (unsigned long long)(uintptr_t)(wdown + (size_t)(m0 + ar) * N_DOWN + k0 + ak);
    asm volatile("global_load_async_to_lds_b128 %0, %1, off\n\t"
                 "global_load_async_to_lds_b128 %0, %1, off offset:16"
                 :: "v"(lds0), "v"(g0) : "memory");
  };
  auto load_b = [&](int k0) {
    const float* bp = loraB + (size_t)(k0 + kseg) * C_DIM + n0 + nB;
#pragma unroll
    for (int i = 0; i < 16; ++i) bv[i] = bp[(size_t)i * C_DIM];
  };
  auto store_b = [&](int buf) {
    unsigned ub[8];
#pragma unroll
    for (int i = 0; i < 8; ++i) ub[i] = pk2_bf16(bv[2 * i], bv[2 * i + 1]);
    *(uint4*)(&Bs[buf][nB * LDB + kseg])     = ((uint4*)ub)[0];
    *(uint4*)(&Bs[buf][nB * LDB + kseg + 8]) = ((uint4*)ub)[1];
  };
  auto compute = [&](int buf) {
    v16bf afr[2], bfr[4];
#pragma unroll
    for (int mi = 0; mi < 2; ++mi) {
      const bf16_t* p = &As[buf][(arow + mi * 16) * LDA + akoff];
      afr[mi] = frag16(p, p + 16);
    }
#pragma unroll
    for (int ni = 0; ni < 4; ++ni) {
      const bf16_t* p = &Bs[buf][(bcol + ni * 16) * LDB + bkoff];
      bfr[ni] = frag16(p, p + 16);
    }
#pragma unroll
    for (int mi = 0; mi < 2; ++mi)
#pragma unroll
      for (int ni = 0; ni < 4; ++ni)
        acc[mi][ni] = __builtin_amdgcn_wmma_f32_16x16x32_bf16(
            false, afr[mi], false, bfr[ni], (short)0, acc[mi][ni], false, false);
  };

  const int NIT = N_DOWN / TK;
  issue_a_async(0, 0);
  load_b(0);
  store_b(0);
  asm volatile("s_wait_asynccnt 0x0" ::: "memory");
  __syncthreads();
#pragma unroll 1
  for (int it = 0; it < NIT; ++it) {
    const int buf = it & 1;
    if (it + 1 < NIT) {
      issue_a_async((it + 1) * TK, buf ^ 1);      // async prefetch, no VGPR staging
      load_b((it + 1) * TK);
    }
    compute(buf);
    if (it + 1 < NIT) {
      store_b(buf ^ 1);
      asm volatile("s_wait_asynccnt 0x0" ::: "memory");
      __syncthreads();
    }
  }

#pragma unroll
  for (int mi = 0; mi < 2; ++mi) {
    const int rbase = m0 + wm * 32 + mi * 16 + ((lane >> 4) << 3);
    const int colb  = n0 + wn * 64 + (lane & 15);
#pragma unroll
    for (int ni = 0; ni < 4; ++ni) {
      const int col = colb + ni * 16;
#pragma unroll
      for (int i = 0; i < 8; ++i)
        out[(size_t)(rbase + i) * C_DIM + col] = SCALING * acc[mi][ni][i];
    }
  }
}

// ---------------------------------------------------------------------------
extern "C" void kernel_launch(void* const* d_in, const int* in_sizes, int n_in,
                              void* d_out, int out_size, void* d_ws, size_t ws_size,
                              hipStream_t stream) {
  const float* x  = (const float*)d_in[0];   // [8,4096,1024]
  const float* gw = (const float*)d_in[1];   // [1024,8]
  const float* gb = (const float*)d_in[2];   // [8]
  const float* lA = (const float*)d_in[3];   // [8,1024,64]
  const float* lB = (const float*)d_in[4];   // [8,64,1024] == [512,1024]
  float* out = (float*)d_out;

  float* w_ws = (float*)d_ws;                                                  // 1 MB
  unsigned short* wdown_st = (unsigned short*)((char*)d_ws + (size_t)T_TOK * E_EXP * 4);
  const bf16_t*   wdown_ld = (const bf16_t*)wdown_st;                          // 32 MB

  smoe_gate<<<T_TOK / 8, 256, 0, stream>>>(x, gw, gb, w_ws);
  smoe_down<<<dim3(N_DOWN / TN, T_TOK / TM), 256, 0, stream>>>(x, lA, w_ws, wdown_st);
  smoe_up<<<dim3(C_DIM / TN, T_TOK / TM), 256, 0, stream>>>(wdown_ld, lB, out);
}
